// EHRModel_28406913696393
// MI455X (gfx1250) — compile-verified, matrix operand order checked by reading, not run on hardware
//
#include <hip/hip_runtime.h>

// ---------------------------------------------------------------------------
// 2-layer LSTM (B=512, T=512, D=76, H=256) + dense head on gfx1250.
//
// Single persistent kernel: block owns 32 batch rows (2 M-tiles) for the
// ENTIRE time loop (LSTM recurrence never crosses batch rows -> no grid
// sync). 32 waves/block = 2 batch-tiles x 16 unit-tiles; each wave computes
// a 16x16 (batch x unit) tile of all four gates via v_wmma_f32_16x16x32_bf16.
// h1/h2 ping-pong tiles live in LDS; c1/c2 live in registers. Weights are
// re-streamed from L2/WGP$ every step (opaque scalar offset stops LICM+spill
// while keeping global_load addressing).
// ---------------------------------------------------------------------------

typedef __bf16  bf16;
typedef bf16    v16bf __attribute__((ext_vector_type(16)));
typedef bf16    v8bf  __attribute__((ext_vector_type(8)));
typedef float   v8f   __attribute__((ext_vector_type(8)));

#define B_      512
#define T_      512
#define D_      76
#define DPAD    96      // 76 padded up to a multiple of 32 (K granularity)
#define H_      256
#define G4H     1024    // 4*H

// ---------------------------------------------------------------------------
// Fragment load per the CDNA5 16-bit A-matrix 16x32 layout:
//   lanes 0-15  : M = lane,    VGPR0-3 = K[k0..k0+7],  VGPR4-7 = K[k0+16..k0+23]
//   lanes 16-31 : M = lane-16, VGPR0-3 = K[k0+8..+15], VGPR4-7 = K[k0+24..+31]
// B fragments use the symmetric pattern on W^T stored [N, K] row-major.
// Works for global or LDS base pointers (addrspace inferred after inlining).
// ---------------------------------------------------------------------------
__device__ __forceinline__ v16bf load_frag(const bf16* tile_base, int ld,
                                           int k0, int lane) {
    const int r    = lane & 15;
    const int koff = k0 + ((lane >> 4) << 3);
    const bf16* p  = tile_base + r * ld + koff;
    v8bf lo = *(const v8bf*)(p);
    v8bf hi = *(const v8bf*)(p + 16);
    return __builtin_shufflevector(lo, hi, 0,1,2,3,4,5,6,7,8,9,10,11,12,13,14,15);
}

__device__ __forceinline__ float sigmoidf_(float x) {
    return 1.0f / (1.0f + __expf(-x));
}

// One K-chunk of A against the four gate column-blocks of W^T.
// All four B fragments are fetched first so the loads clause together and
// stay in flight, then the four WMMAs issue back-to-back.
__device__ __forceinline__ void accum_gates(v8f acc[4], v16bf a,
                                            const bf16* __restrict__ WT, int ld,
                                            int k0, int ut, int lane) {
    v16bf b[4];
    #pragma unroll
    for (int g = 0; g < 4; ++g)
        b[g] = load_frag(WT + ((g << 8) + (ut << 4)) * ld, ld, k0, lane);
    #pragma unroll
    for (int g = 0; g < 4; ++g)
        acc[g] = __builtin_amdgcn_wmma_f32_16x16x32_bf16(
            false, a, false, b[g], (short)0, acc[g], false, false);
}

// Lane-local gate math + cell update + h store into an LDS tile [16 x 256].
__device__ __forceinline__ void gate_update(v8f acc[4], const float bz[4],
                                            v8f& c, bf16* hout,
                                            int ucol, int lane) {
    const int mhi = (lane >> 4) << 3;
    #pragma unroll
    for (int r = 0; r < 8; ++r) {
        float i = sigmoidf_(acc[0][r] + bz[0]);
        float f = sigmoidf_(acc[1][r] + bz[1]);
        float g = tanhf    (acc[2][r] + bz[2]);
        float o = sigmoidf_(acc[3][r] + bz[3]);
        float cn = f * c[r] + i * g;
        c[r] = cn;
        hout[(r + mhi) * H_ + ucol] = (bf16)(o * tanhf(cn));
    }
}

// ---------------------------------------------------------------------------
// Persistent kernel: 16 blocks x 1024 threads (32 waves = 2 batch-tiles x
// 16 unit-tiles). Runs both LSTM layers over all T timesteps for its 32
// batch rows, then the dense head.
// ---------------------------------------------------------------------------
__global__ __launch_bounds__(1024)
void lstm_persistent(const bf16*  __restrict__ xT,    // [T, 512, 96]  bf16
                     const bf16*  __restrict__ WiT1,  // [1024, 96]
                     const bf16*  __restrict__ WhT1,  // [1024, 256]
                     const bf16*  __restrict__ WiT2,  // [1024, 256]
                     const bf16*  __restrict__ WhT2,  // [1024, 256]
                     const float* __restrict__ b1,    // [1024]
                     const float* __restrict__ b2,    // [1024]
                     const float* __restrict__ Wd,    // [256, 25]
                     const float* __restrict__ bd,    // [25]
                     float*       __restrict__ out)   // [512, 25]
{
    __shared__ bf16 h1s[2][2][16 * H_];   // [pingpong][batch-subtile][16x256]
    __shared__ bf16 h2s[2][2][16 * H_];

    const int wave = threadIdx.x >> 5;    // 0..31
    const int lane = threadIdx.x & 31;
    const int btl  = wave >> 4;           // batch subtile in block: 0..1
    const int ut   = wave & 15;           // unit tile 0..15
    const int bt   = (blockIdx.x << 1) + btl;      // global batch tile 0..31
    const int ucol = (ut << 4) + (lane & 15);      // hidden unit 0..255

    // zero-init hidden state tiles (both buffers, as 32-bit stores)
    {
        int* z0 = (int*)&h1s[0][0][0];
        int* z1 = (int*)&h2s[0][0][0];
        const int n = 2 * 2 * 16 * H_ / 2;
        for (int i = threadIdx.x; i < n; i += 1024) { z0[i] = 0; z1[i] = 0; }
    }
    __syncthreads();

    // biases for this wave's unit column (constant over time)
    const float bz1[4] = { b1[ucol], b1[H_ + ucol], b1[2*H_ + ucol], b1[3*H_ + ucol] };
    const float bz2[4] = { b2[ucol], b2[H_ + ucol], b2[2*H_ + ucol], b2[3*H_ + ucol] };

    v8f c1 = {};   // layer-1 cell state fragment (registers, whole run)
    v8f c2 = {};   // layer-2 cell state fragment

    const bf16* Xbase = xT + (size_t)(bt << 4) * DPAD;

    // Opaque element offset, redefined every iteration: defeats hoisting of
    // the loop-invariant weight fragments (which would spill to scratch),
    // while the weight pointers themselves remain traceable to global
    // kernargs so the loads keep the global_load/saddr form (LOADcnt only,
    // no FLAT/DScnt coupling).
    int wofs = 0;

    #pragma unroll 1
    for (int t = 0; t < T_; ++t) {
        asm volatile("" : "+s"(wofs));
        const bf16* w_i1 = WiT1 + wofs;
        const bf16* w_h1 = WhT1 + wofs;
        const bf16* w_i2 = WiT2 + wofs;
        const bf16* w_h2 = WhT2 + wofs;

        const int pi = t & 1, ni = (t + 1) & 1;
        const bf16* Xt = Xbase + (size_t)t * B_ * DPAD;

        // prefetch next timestep's x rows (global_prefetch path)
        if (t + 1 < T_)
            __builtin_prefetch(Xt + (size_t)B_ * DPAD + (lane & 15) * DPAD, 0, 1);

        // ================= layer 1 =================
        {
            v8f acc[4] = {};
            #pragma unroll
            for (int k0 = 0; k0 < DPAD; k0 += 32) {          // x @ Wi1
                v16bf a = load_frag(Xt, DPAD, k0, lane);
                accum_gates(acc, a, w_i1, DPAD, k0, ut, lane);
            }
            #pragma unroll
            for (int k0 = 0; k0 < H_; k0 += 32) {            // h1_prev @ Wh1
                v16bf a = load_frag(&h1s[pi][btl][0], H_, k0, lane);
                accum_gates(acc, a, w_h1, H_, k0, ut, lane);
            }
            gate_update(acc, bz1, c1, &h1s[ni][btl][0], ucol, lane);
        }
        __syncthreads();   // h1[ni] visible to all waves of this batch tile

        // ================= layer 2 =================
        {
            v8f acc[4] = {};
            #pragma unroll
            for (int k0 = 0; k0 < H_; k0 += 32) {            // h1_t @ Wi2
                v16bf a = load_frag(&h1s[ni][btl][0], H_, k0, lane);
                accum_gates(acc, a, w_i2, H_, k0, ut, lane);
            }
            #pragma unroll
            for (int k0 = 0; k0 < H_; k0 += 32) {            // h2_prev @ Wh2
                v16bf a = load_frag(&h2s[pi][btl][0], H_, k0, lane);
                accum_gates(acc, a, w_h2, H_, k0, ut, lane);
            }
            gate_update(acc, bz2, c2, &h2s[ni][btl][0], ucol, lane);
        }
        __syncthreads();   // h2[ni] visible; also fences ping-pong reuse
    }

    // ---- dense head: logits[32,25] = h2_final @ Wd + bd ----
    // T even -> final h2 lives in buffer 0. h2 tiles are in LDS already.
    for (int idx = threadIdx.x; idx < 32 * 25; idx += 1024) {
        const int r = idx / 25, j = idx % 25;
        const int sub = r >> 4, rr = r & 15;
        float s = bd[j];
        #pragma unroll 4
        for (int k = 0; k < H_; ++k)
            s += (float)h2s[0][sub][rr * H_ + k] * Wd[k * 25 + j];
        out[((blockIdx.x << 5) + r) * 25 + j] = s;
    }
}

// ---------------------------------------------------------------------------
// Prep kernels
// ---------------------------------------------------------------------------

// x [B, T, 76] fp32  ->  xT [T, B, 96] bf16 (zero-padded K), time-major slabs
__global__ void repack_x(const float* __restrict__ x, bf16* __restrict__ xT, int total) {
    int idx = blockIdx.x * blockDim.x + threadIdx.x;
    if (idx >= total) return;
    int k  = idx % DPAD;
    int tb = idx / DPAD;
    int b  = tb % B_;
    int t  = tb / B_;
    float v = (k < D_) ? x[(b * T_ + t) * D_ + k] : 0.0f;
    xT[idx] = (bf16)v;
}

// W [K, 1024] fp32 -> W^T [1024, Kpad] bf16 (zero-padded K)
__global__ void transpose_w(const float* __restrict__ W, bf16* __restrict__ WT,
                            int K, int Kpad) {
    int idx = blockIdx.x * blockDim.x + threadIdx.x;
    if (idx >= G4H * Kpad) return;
    int k = idx % Kpad;
    int n = idx / Kpad;
    float v = (k < K) ? W[k * G4H + n] : 0.0f;
    WT[idx] = (bf16)v;
}

// ---------------------------------------------------------------------------
// Host launcher
// ---------------------------------------------------------------------------
extern "C" void kernel_launch(void* const* d_in, const int* in_sizes, int n_in,
                              void* d_out, int out_size, void* d_ws, size_t ws_size,
                              hipStream_t stream) {
    const float* x   = (const float*)d_in[0];   // [512, 512, 76]
    const float* Wi1 = (const float*)d_in[1];   // [76, 1024]
    const float* Wh1 = (const float*)d_in[2];   // [256, 1024]
    const float* b1  = (const float*)d_in[3];   // [1024]
    const float* Wi2 = (const float*)d_in[4];   // [256, 1024]
    const float* Wh2 = (const float*)d_in[5];   // [256, 1024]
    const float* b2  = (const float*)d_in[6];   // [1024]
    const float* Wd  = (const float*)d_in[7];   // [256, 25]
    const float* bd  = (const float*)d_in[8];   // [25]
    float* out = (float*)d_out;                 // [512, 25]

    // ---- workspace carve-up ----
    char* p = (char*)d_ws;
    size_t off = 0;
    auto take = [&](size_t bytes) { char* q = p + off; off += (bytes + 255) & ~(size_t)255; return q; };

    bf16* xT   = (bf16*)take((size_t)T_ * B_ * DPAD * 2);   // 48 MB
    bf16* WiT1 = (bf16*)take((size_t)G4H * DPAD * 2);
    bf16* WhT1 = (bf16*)take((size_t)G4H * H_   * 2);
    bf16* WiT2 = (bf16*)take((size_t)G4H * H_   * 2);
    bf16* WhT2 = (bf16*)take((size_t)G4H * H_   * 2);

    // ---- prep: weight transposes + x repack ----
    {
        int n, blk = 256;
        n = G4H * DPAD; transpose_w<<<(n+blk-1)/blk, blk, 0, stream>>>(Wi1, WiT1, D_, DPAD);
        n = G4H * H_;   transpose_w<<<(n+blk-1)/blk, blk, 0, stream>>>(Wh1, WhT1, H_, H_);
        n = G4H * H_;   transpose_w<<<(n+blk-1)/blk, blk, 0, stream>>>(Wi2, WiT2, H_, H_);
        n = G4H * H_;   transpose_w<<<(n+blk-1)/blk, blk, 0, stream>>>(Wh2, WhT2, H_, H_);
        n = T_ * B_ * DPAD;
        repack_x<<<(n+blk-1)/blk, blk, 0, stream>>>(x, xT, n);
    }

    // ---- persistent recurrent kernel (both layers, all timesteps, head) ----
    lstm_persistent<<<16, 1024, 0, stream>>>(xT, WiT1, WhT1, WiT2, WhT2,
                                             b1, b2, Wd, bd, out);
}
